// GMMINR_58823872086171
// MI455X (gfx1250) — matrix operand order repaired and checked by reading, not compiled, not run on hardware
//
#include <hip/hip_runtime.h>
#include <hip/hip_bf16.h>

typedef __attribute__((ext_vector_type(16))) _Float16 v16h;
typedef __attribute__((ext_vector_type(8)))  _Float16 v8h;
typedef __attribute__((ext_vector_type(8)))  float    v8f;

#define NPTS    262144
#define GNUM    128
#define FNUM    128
#define HDIM    512
#define IN0     176
#define K1PAD   192
#define MTILE   64
#define NTHREADS 256

// workspace layout (in halfs)
#define OFF_GFT  0
#define OFF_W1P  (OFF_GFT + GNUM*FNUM)        // 16384
#define OFF_W2   (OFF_W1P + HDIM*K1PAD)       // 114688
#define OFF_W3   (OFF_W2  + HDIM*HDIM)
#define OFF_W4   (OFF_W3  + HDIM*HDIM)

// LDS layout (bytes); all row strides padded +8 halfs (16B) -> 4-bank shift/row
#define LDS_SG   0                 // float sG[128][8]; reused as sRed[64][4] at the end
#define LDS_IN   4096              // _Float16 sIn[64][200]
#define LDS_X    29696             // _Float16 sX [64][520]  (also dens[64][136])
#define LDS_Y    96256             // _Float16 sY [64][520]
#define LDS_TOTAL 162816

// ---- fragment loaders (16-bit WMMA operands, contiguous k per lane half, wave32) ----

// A 16x32 f16: lanes {0-15,16-31} hold M=lane&15; per lane 16 contiguous
// halfs at k = kb + (hi?16:0) .. +15.
__device__ inline v16h load_a_frag(const _Float16* buf, int ld, int r, int kb, int lane) {
    __builtin_assume(buf != nullptr);
    int off = (lane & 16) ? 16 : 0;
    const _Float16* p = buf + r * ld + kb + off;
    v8h lo = *(const v8h*)p;
    v8h hi = *(const v8h*)(p + 8);
    return __builtin_shufflevector(lo, hi, 0,1,2,3,4,5,6,7,8,9,10,11,12,13,14,15);
}

#define WMMA_F16(a, b, c) __builtin_amdgcn_wmma_f32_16x16x32_f16( \
    false, (a), false, (b), (short)0, (c), false, false)

// ACT: 0 = identity (no bias), 1 = leaky_relu(0.2), 2 = sin(30x)
// Wave partition: 2 M-groups x 4 N-groups; each wave owns 2 row tiles and
// SPW column strips. A fragments register-cached across all strips
// (LDS A-traffic = 1KB/SPW per wmma). B fragments explicitly
// double-buffered across k so ~2*SPW loads stay in flight (hides L2 latency).
template<int ACT, int SPW>
__device__ inline void layer_wmma(const _Float16* __restrict__ Wh, int ldw,
                                  const float* __restrict__ bias,
                                  const _Float16* in, int ldin, int K,
                                  _Float16* outb, int ldout,
                                  int wave, int lane)
{
    __builtin_assume(in != nullptr);
    __builtin_assume(outb != nullptr);
    const int mg = wave & 1;    // row-tile group: tiles {2mg, 2mg+1}
    const int ng = wave >> 1;   // column group: strips ng*SPW .. ng*SPW+SPW-1

    v8f acc[2][SPW];
    #pragma unroll
    for (int t = 0; t < 2; ++t)
        #pragma unroll
        for (int j = 0; j < SPW; ++j)
            #pragma unroll
            for (int i = 0; i < 8; ++i) acc[t][j][i] = 0.f;

    const int r0 = (mg * 2) * 16 + (lane & 15);
    // Base pointer for this lane's B rows: column o = strip*16 + (lane&15),
    // k-base (lane>=16 ? 16 : 0). Strip deltas are compile-time constants
    // (16*ldw halfs) and fold into load-immediate offsets.
    const _Float16* wbase = Wh + ((ng * SPW * 16 + (lane & 15)) * ldw)
                               + ((lane & 16) ? 16 : 0);

    v16h bc[SPW], bn[SPW];
    #pragma unroll
    for (int j = 0; j < SPW; ++j)
        bc[j] = *(const v16h*)(wbase + j * 16 * ldw);

    for (int kb = 0; kb < K; kb += 64) {
        // ---- phase 0: prefetch k=kb+32, compute k=kb with bc ----
        int kp1 = (kb + 32 < K) ? (kb + 32) : kb;
        #pragma unroll
        for (int j = 0; j < SPW; ++j)
            bn[j] = *(const v16h*)(wbase + j * 16 * ldw + kp1);
        {
            v16h a0 = load_a_frag(in, ldin, r0,      kb, lane);
            v16h a1 = load_a_frag(in, ldin, r0 + 16, kb, lane);
            #pragma unroll
            for (int j = 0; j < SPW; ++j) {
                acc[0][j] = WMMA_F16(a0, bc[j], acc[0][j]);
                acc[1][j] = WMMA_F16(a1, bc[j], acc[1][j]);
            }
        }
        // ---- phase 1: prefetch k=kb+64, compute k=kb+32 with bn ----
        if (kb + 32 < K) {
            int kp2 = (kb + 64 < K) ? (kb + 64) : kp1;
            #pragma unroll
            for (int j = 0; j < SPW; ++j)
                bc[j] = *(const v16h*)(wbase + j * 16 * ldw + kp2);
            v16h a0 = load_a_frag(in, ldin, r0,      kb + 32, lane);
            v16h a1 = load_a_frag(in, ldin, r0 + 16, kb + 32, lane);
            #pragma unroll
            for (int j = 0; j < SPW; ++j) {
                acc[0][j] = WMMA_F16(a0, bn[j], acc[0][j]);
                acc[1][j] = WMMA_F16(a1, bn[j], acc[1][j]);
            }
        }
    }

    const int hi8 = (lane & 16) ? 8 : 0;
    #pragma unroll
    for (int j = 0; j < SPW; ++j) {
        int col = (ng * SPW + j) * 16 + (lane & 15);
        float bv = (ACT != 0) ? bias[col] : 0.f;   // compile-time branch
        #pragma unroll
        for (int t = 0; t < 2; ++t) {
            int rowb = (mg * 2 + t) * 16 + hi8;
            #pragma unroll
            for (int i = 0; i < 8; ++i) {
                float v = acc[t][j][i] + bv;
                if (ACT == 1) v = v > 0.f ? v : 0.2f * v;
                else if (ACT == 2) v = __sinf(30.f * v);
                outb[(rowb + i) * ldout + col] = (_Float16)v;
            }
        }
    }
}

__global__ void __launch_bounds__(NTHREADS)
gmm_inr_kernel(const float* __restrict__ x,
               const float* __restrict__ centers,
               const float* __restrict__ cov,
               const float* __restrict__ b1,
               const float* __restrict__ b2,
               const float* __restrict__ b3,
               const float* __restrict__ b4,
               const float* __restrict__ W5,
               const float* __restrict__ b5,
               const _Float16* __restrict__ gfT,
               const _Float16* __restrict__ W1p,
               const _Float16* __restrict__ W2h,
               const _Float16* __restrict__ W3h,
               const _Float16* __restrict__ W4h,
               float* __restrict__ out)
{
    extern __shared__ char smem[];
    char* smb = smem;
    __builtin_assume(smb != nullptr);
    float*    sG  = (float*)(smb + LDS_SG);       // [128][8] cx,cy,cz,i0,i1,i2,coeff,pad
    _Float16* sIn = (_Float16*)(smb + LDS_IN);    // [64][200] feat(0..127) ++ pe(128..175) ++ zero pad
    _Float16* sX  = (_Float16*)(smb + LDS_X);     // [64][520]
    _Float16* sY  = (_Float16*)(smb + LDS_Y);     // [64][520]
    _Float16* sD  = sX;                           // dens [64][136] (overwritten by layer1 output)

    const int tid  = threadIdx.x;
    const int lane = tid & 31;
    const int wave = tid >> 5;
    const int rowBase = blockIdx.x * MTILE;

    // ---- per-gaussian constants (diag cov -> inv diag, det coeff) ----
    if (tid < GNUM) {
        int g = tid;
        float d0 = cov[g*9 + 0], d1 = cov[g*9 + 4], d2 = cov[g*9 + 8];
        sG[g*8 + 0] = centers[g*3 + 0];
        sG[g*8 + 1] = centers[g*3 + 1];
        sG[g*8 + 2] = centers[g*3 + 2];
        sG[g*8 + 3] = 1.f / d0;
        sG[g*8 + 4] = 1.f / d1;
        sG[g*8 + 5] = 1.f / d2;
        sG[g*8 + 6] = 0.06349363593424097f * rsqrtf(d0 * d1 * d2); // (2pi)^-1.5/sqrt(det)
        sG[g*8 + 7] = 0.f;
    }
    __syncthreads();

    // ---- density + positional encoding ----
    {
        int r = tid & 63;
        int q = tid >> 6;          // 4 threads per row
        int n = rowBase + r;
        float px = x[n*3 + 0], py = x[n*3 + 1], pz = x[n*3 + 2];
        for (int g = q * 32; g < q * 32 + 32; ++g) {
            float dx = px - sG[g*8+0], dy = py - sG[g*8+1], dz = pz - sG[g*8+2];
            float m = dx*dx*sG[g*8+3] + dy*dy*sG[g*8+4] + dz*dz*sG[g*8+5];
            sD[r * 136 + g] = (_Float16)(sG[g*8+6] * __expf(-0.5f * m));
        }
        float p3[3] = {px, py, pz};
        for (int c = q * 12; c < q * 12 + 12; ++c) {
            int l = c / 6, rr = c % 6;
            float freq = 3.14159265358979f * (float)(1 << l);
            float v = p3[rr < 3 ? rr : rr - 3] * freq;
            sIn[r * 200 + 128 + c] = (_Float16)(rr < 3 ? __sinf(v) : __cosf(v));
        }
        if (q == 0)
            for (int c = IN0; c < K1PAD; ++c) sIn[r * 200 + c] = (_Float16)0.f;
    }
    __syncthreads();

    // ---- feat = dens @ gfeat -> sIn[:, 0:128]  (128 cols -> SPW=2) ----
    layer_wmma<0, 2>(gfT, GNUM, nullptr, sD, 136, 128, sIn, 200, wave, lane);
    __syncthreads();
    // ---- layer 1: [64,192] @ W1p.T -> leaky_relu -> sX ----
    layer_wmma<1, 8>(W1p, K1PAD, b1, sIn, 200, K1PAD, sX, 520, wave, lane);
    __syncthreads();
    // ---- layers 2..4: sin(30x), 512x512 ----
    layer_wmma<2, 8>(W2h, HDIM, b2, sX, 520, HDIM, sY, 520, wave, lane);
    __syncthreads();
    layer_wmma<2, 8>(W3h, HDIM, b3, sY, 520, HDIM, sX, 520, wave, lane);
    __syncthreads();
    layer_wmma<2, 8>(W4h, HDIM, b4, sX, 520, HDIM, sY, 520, wave, lane);
    __syncthreads();

    // ---- final 512 -> 1: all 256 threads, 4-way partial + LDS reduce ----
    {
        float* sRed = sG;          // reuse constant area as [64][4] partials
        int r = tid & 63;
        int q = tid >> 6;
        const _Float16* rowp = sY + r * 520 + q * 128;
        const float* w = W5 + q * 128;
        float a = 0.f;
        for (int h = 0; h < 128; ++h) a += (float)rowp[h] * w[h];
        __syncthreads();           // sG constants no longer needed
        sRed[r * 4 + q] = a;
        __syncthreads();
        if (tid < MTILE)
            out[rowBase + tid] = sRed[tid*4+0] + sRed[tid*4+1]
                               + sRed[tid*4+2] + sRed[tid*4+3] + b5[0];
    }
}

// ---- weight conversion kernels (fp32 -> fp16 in workspace) ----
__global__ void k_cvt(const float* __restrict__ src, _Float16* __restrict__ dst, int n) {
    int i = blockIdx.x * blockDim.x + threadIdx.x;
    if (i < n) dst[i] = (_Float16)src[i];
}
__global__ void k_cvt_w1p(const float* __restrict__ W1, _Float16* __restrict__ dst) {
    int i = blockIdx.x * blockDim.x + threadIdx.x;
    if (i < HDIM * K1PAD) {
        int o = i / K1PAD, k = i % K1PAD;
        dst[i] = (k < IN0) ? (_Float16)W1[o * IN0 + k] : (_Float16)0.f;
    }
}
__global__ void k_cvt_gfT(const float* __restrict__ gf, _Float16* __restrict__ dst) {
    int i = blockIdx.x * blockDim.x + threadIdx.x;
    if (i < GNUM * FNUM) {
        int f = i / GNUM, g = i % GNUM;
        dst[i] = (_Float16)gf[g * FNUM + f];   // gfeat^T so B fragments are K-contiguous
    }
}

extern "C" void kernel_launch(void* const* d_in, const int* in_sizes, int n_in,
                              void* d_out, int out_size, void* d_ws, size_t ws_size,
                              hipStream_t stream) {
    const float* x       = (const float*)d_in[0];
    const float* centers = (const float*)d_in[1];
    const float* cov     = (const float*)d_in[2];
    const float* gfeat   = (const float*)d_in[3];
    const float* W1      = (const float*)d_in[4];
    const float* b1      = (const float*)d_in[5];
    const float* W2      = (const float*)d_in[6];
    const float* b2      = (const float*)d_in[7];
    const float* W3      = (const float*)d_in[8];
    const float* b3      = (const float*)d_in[9];
    const float* W4      = (const float*)d_in[10];
    const float* b4      = (const float*)d_in[11];
    const float* W5      = (const float*)d_in[12];
    const float* b5      = (const float*)d_in[13];

    _Float16* ws  = (_Float16*)d_ws;
    _Float16* gfT = ws + OFF_GFT;
    _Float16* W1p = ws + OFF_W1P;
    _Float16* W2h = ws + OFF_W2;
    _Float16* W3h = ws + OFF_W3;
    _Float16* W4h = ws + OFF_W4;

    k_cvt_gfT<<<(GNUM * FNUM + 255) / 256, 256, 0, stream>>>(gfeat, gfT);
    k_cvt_w1p<<<(HDIM * K1PAD + 255) / 256, 256, 0, stream>>>(W1, W1p);
    k_cvt<<<(HDIM * HDIM + 255) / 256, 256, 0, stream>>>(W2, W2h, HDIM * HDIM);
    k_cvt<<<(HDIM * HDIM + 255) / 256, 256, 0, stream>>>(W3, W3h, HDIM * HDIM);
    k_cvt<<<(HDIM * HDIM + 255) / 256, 256, 0, stream>>>(W4, W4h, HDIM * HDIM);

    gmm_inr_kernel<<<NPTS / MTILE, NTHREADS, LDS_TOTAL, stream>>>(
        x, centers, cov, b1, b2, b3, b4, W5, b5,
        gfT, W1p, W2h, W3h, W4h, (float*)d_out);
}
